// PraxisAttention_12283606466579
// MI455X (gfx1250) — matrix operand order repaired and checked by reading, not compiled, not run on hardware
//
#include <hip/hip_runtime.h>
#include <hip/hip_bf16.h>
#include <math.h>

typedef __attribute__((ext_vector_type(16))) __bf16 v16bf;
typedef __attribute__((ext_vector_type(8)))  float  v8f;
typedef __attribute__((ext_vector_type(4)))  unsigned int u32x4;
typedef __attribute__((ext_vector_type(8)))  int i32x8;
typedef __attribute__((ext_vector_type(4)))  int i32x4;

#define BB 2
#define SS 2048
#define DD 1024
#define HH 16
#define HDIM 64
#define NROW (BB*SS)          /* 4096 */
#define QK_LD (2*DD)          /* 2048 */
#define LAMBDA_INIT 0.8f

#if defined(__gfx1250__) && __has_builtin(__builtin_amdgcn_tensor_load_to_lds)
#define HAVE_TDM 1
#if __has_include(<hip/amd_detail/amd_gfx1250_TDM.h>)
#define TDM_SIX_ARGS 1
#endif
#else
#define HAVE_TDM 0
#endif

static __device__ __forceinline__ v8f wmma_bf16(v16bf a, v16bf b, v8f c) {
  // 8 args: (neg_a, A, neg_b, B, c_mod, C, reuse_a, reuse_b)
  return __builtin_amdgcn_wmma_f32_16x16x32_bf16(false, a, false, b, (short)0, c, false, false);
}

static __device__ __forceinline__ v8f vzero8() {
  v8f z;
#pragma unroll
  for (int i = 0; i < 8; ++i) z[i] = 0.0f;
  return z;
}

// ---------------------------------------------------------------------------
// Fragment loaders (bf16 WMMA, wave32 layouts per CDNA5 ISA 7.12.2)
//   A (16x32, MxK): lane<16 -> row=lane,   K in {0..7, 16..23}
//                   lane>=16 -> row=lane-16, K in {8..15, 24..31}
//   B (32x16, KxN): lane -> col = lane&15,  K same as A: {0..7,16..23} / {8..15,24..31}
//   C/D (16x16):    lane -> col = lane&15,  row = r + (lane>=16?8:0)
// _h variants read native bf16 (no conversions in the hot loop).
// ---------------------------------------------------------------------------

static __device__ __forceinline__ v16bf load_a_frag_h(const __bf16* __restrict__ M,
                                                      int row0, int k0, int ld) {
  const int lane = threadIdx.x & 31;
  const int hi = lane >> 4;
  const __bf16* p = M + (size_t)(row0 + (lane & 15)) * ld + k0 + hi * 8;
  v16bf a;
#pragma unroll
  for (int e = 0; e < 8; ++e) { a[e] = p[e]; a[8 + e] = p[16 + e]; }
  return a;
}

static __device__ __forceinline__ v16bf load_b_frag_h(const __bf16* __restrict__ M,
                                                      int k0, int col0, int ld) {
  const int lane = threadIdx.x & 31;
  const int kb = k0 + ((lane >> 4) << 3);
  const int col = col0 + (lane & 15);
  v16bf b;
#pragma unroll
  for (int e = 0; e < 8; ++e) { b[e] = M[(size_t)(kb + e) * ld + col]; b[8 + e] = M[(size_t)(kb + 16 + e) * ld + col]; }
  return b;
}

// B fragment where the contraction dim is CONTIGUOUS in memory.
static __device__ __forceinline__ v16bf load_bT_frag_h(const __bf16* __restrict__ M,
                                                       int n0, int k0, int ld) {
  const int lane = threadIdx.x & 31;
  const __bf16* p = M + (size_t)(n0 + (lane & 15)) * ld + k0 + ((lane >> 4) << 3);
  v16bf b;
#pragma unroll
  for (int e = 0; e < 8; ++e) { b[e] = p[e]; b[8 + e] = p[16 + e]; }
  return b;
}

// f32-source B fragment (small matrices: mem state)
static __device__ __forceinline__ v16bf load_b_frag_f(const float* __restrict__ M,
                                                      int k0, int col0, int ld) {
  const int lane = threadIdx.x & 31;
  const int kb = k0 + ((lane >> 4) << 3);
  const int col = col0 + (lane & 15);
  v16bf b;
#pragma unroll
  for (int e = 0; e < 8; ++e) { b[e] = (__bf16)M[(size_t)(kb + e) * ld + col]; b[8 + e] = (__bf16)M[(size_t)(kb + 16 + e) * ld + col]; }
  return b;
}

static __device__ __forceinline__ void store_c_f32(float* __restrict__ M, v8f c,
                                                   int row0, int col0, int ld) {
  const int lane = threadIdx.x & 31;
  const int rb = row0 + ((lane >> 4) << 3);
  const int col = col0 + (lane & 15);
#pragma unroll
  for (int r = 0; r < 8; ++r) M[(size_t)(rb + r) * ld + col] = c[r];
}

static __device__ __forceinline__ void store_c_bf16(__bf16* __restrict__ M, v8f c,
                                                    int row0, int col0, int ld) {
  const int lane = threadIdx.x & 31;
  const int rb = row0 + ((lane >> 4) << 3);
  const int col = col0 + (lane & 15);
#pragma unroll
  for (int r = 0; r < 8; ++r) M[(size_t)(rb + r) * ld + col] = (__bf16)c[r];
}

// ---------------------------------------------------------------------------
// Tensor Data Mover: 2-D tile load (global -> LDS), D# per ISA 08 §8.
// dsz_code: 0=1B 1=2B 2=4B 3=8B; all dims/strides in data_size units.
// ---------------------------------------------------------------------------
#if HAVE_TDM
static __device__ __forceinline__ unsigned lds_offset_of(const void* p) {
  return (unsigned)(uintptr_t)(__attribute__((address_space(3))) const char*)p;
}

static __device__ __forceinline__ void tdm_load_tile(
    unsigned lds_byte_off, const void* gbase, unsigned dsz_code,
    unsigned tensor_w, unsigned tensor_h,
    unsigned tile_w, unsigned tile_h, unsigned row_stride_elems) {
  unsigned long long ga = (unsigned long long)(uintptr_t)gbase;
  u32x4 g0;
  g0[0] = 1u;                                       // count=1, user load D#
  g0[1] = lds_byte_off;                             // lds_addr
  g0[2] = (unsigned)ga;                             // global_addr[31:0]
  g0[3] = ((unsigned)(ga >> 32) & 0x01FFFFFFu)      // global_addr[56:32]
          | 0x80000000u;                            // type=2 ("image")
  i32x8 g1;
  g1[0] = (int)(dsz_code << 16);                    // data_size, no multicast
  g1[1] = (int)((tensor_w & 0xFFFFu) << 16);        // tensor_dim0[15:0]
  g1[2] = (int)((tensor_w >> 16) | ((tensor_h & 0xFFFFu) << 16));
  g1[3] = (int)((tensor_h >> 16) | (tile_w << 16)); // tile_dim0
  g1[4] = (int)(tile_h & 0xFFFFu);                  // tile_dim1 (tile_dim2=0)
  g1[5] = (int)row_stride_elems;                    // tensor_dim0_stride[31:0]
  g1[6] = 0;
  g1[7] = 0;
  i32x4 gz;
  gz[0] = 0; gz[1] = 0; gz[2] = 0; gz[3] = 0;
#if defined(TDM_SIX_ARGS)
  i32x8 gz8;
#pragma unroll
  for (int i = 0; i < 8; ++i) gz8[i] = 0;
  __builtin_amdgcn_tensor_load_to_lds(g0, g1, gz, gz, gz8, 0);
#else
  __builtin_amdgcn_tensor_load_to_lds(g0, g1, gz, gz, 0);
#endif
}
#endif

// ---------------------------------------------------------------------------
// GEMM: C[M,N] = A[M,K] @ B[K,N], A/B native bf16 row-major, f32 accumulate.
// TDM path: block computes a 64x64 C tile; wave0 drives a double-buffered TDM
// pipeline (A 64x32 slab + B 32x64 slab per K-step, 4KB each) into LDS; all
// four waves consume bf16 fragments from LDS with zero conversions.
// ---------------------------------------------------------------------------
template <bool BF16_OUT>
__global__ void __launch_bounds__(128)
gemm_wmma_h(const __bf16* __restrict__ A, const __bf16* __restrict__ B,
            void* __restrict__ Cout, int M, int N, int K) {
  const int wave = threadIdx.x >> 5;
  const int rowB = blockIdx.y * 64;
  const int col0 = blockIdx.x * 64;
  const int row0 = rowB + wave * 16;

  v8f acc[4];
#pragma unroll
  for (int f = 0; f < 4; ++f) acc[f] = vzero8();

#if HAVE_TDM
  __shared__ __bf16 Abuf[2][64 * 32];
  __shared__ __bf16 Bbuf[2][32 * 64];

  if (wave == 0) {
    tdm_load_tile(lds_offset_of(&Abuf[0][0]), A + (size_t)rowB * K, 1u,
                  (unsigned)K, (unsigned)M, 32u, 64u, (unsigned)K);
    tdm_load_tile(lds_offset_of(&Bbuf[0][0]), B + col0, 1u,
                  (unsigned)N, (unsigned)K, 64u, 32u, (unsigned)N);
  }

  int buf = 0;
  for (int k0 = 0; k0 < K; k0 += 32) {
    if (wave == 0) {
      if (k0 + 32 < K) {
        tdm_load_tile(lds_offset_of(&Abuf[buf ^ 1][0]),
                      A + (size_t)rowB * K + (k0 + 32), 1u,
                      (unsigned)K, (unsigned)M, 32u, 64u, (unsigned)K);
        tdm_load_tile(lds_offset_of(&Bbuf[buf ^ 1][0]),
                      B + (size_t)(k0 + 32) * N + col0, 1u,
                      (unsigned)N, (unsigned)K, 64u, 32u, (unsigned)N);
        __builtin_amdgcn_s_wait_tensorcnt(2);   // current slab complete
      } else {
        __builtin_amdgcn_s_wait_tensorcnt(0);
      }
    }
    __syncthreads();
    v16bf a = load_a_frag_h(&Abuf[buf][0], wave * 16, 0, 32);
#pragma unroll
    for (int f = 0; f < 4; ++f) {
      v16bf bfr = load_b_frag_h(&Bbuf[buf][0], 0, f * 16, 64);
      acc[f] = wmma_bf16(a, bfr, acc[f]);
    }
    __syncthreads();
    buf ^= 1;
  }
#else
  // Fallback (host pass / toolchains without the TDM builtin)
  if (row0 >= M || col0 >= N) return;
  for (int k0 = 0; k0 < K; k0 += 32) {
    v16bf a = load_a_frag_h(A, row0, k0, K);
#pragma unroll
    for (int f = 0; f < 4; ++f)
      acc[f] = wmma_bf16(a, load_b_frag_h(B, k0, col0 + f * 16, N), acc[f]);
  }
#endif
#pragma unroll
  for (int f = 0; f < 4; ++f) {
    if (BF16_OUT)
      store_c_bf16((__bf16*)Cout, acc[f], row0, col0 + f * 16, N);
    else
      store_c_f32((float*)Cout, acc[f], row0, col0 + f * 16, N);
  }
}

// ---------------------------------------------------------------------------
// Fused differential flash attention + linear-memory term. Q/K/V are bf16.
// Grid (S/64, H, B); 4 waves/block, one 16-row q-tile per wave.
// Scores computed TRANSPOSED (K·Qᵀ) so softmaxed P^T (C-layout) is already
// the A-fragment of P for the P@V WMMA — no cross-lane shuffles for data.
// ---------------------------------------------------------------------------
__global__ void __launch_bounds__(128)
diffattn_kernel(const __bf16* __restrict__ Qb, const __bf16* __restrict__ Kb,
                const __bf16* __restrict__ Vb, const float* __restrict__ amask,
                const float* __restrict__ lam_q1, const float* __restrict__ lam_q2,
                const float* __restrict__ lam_k1, const float* __restrict__ lam_k2,
                const float* __restrict__ mgate, const float* __restrict__ memM,
                const float* __restrict__ zv,
                float* __restrict__ Xout, float* __restrict__ stats) {
  const int lane = threadIdx.x & 31;
  const int wave = threadIdx.x >> 5;
  const int hi = lane >> 4;       // lane half
  const int ln = lane & 15;
  const int b = blockIdx.z, h = blockIdx.y;
  const int q0 = blockIdx.x * 64 + wave * 16;

  const float scale = 0.125f;  // 1/sqrt(64)
  const float slope = exp2f(-8.0f * (float)(h + 1) / (float)HH);

  float d1 = 0.f, d2 = 0.f;
  for (int i = 0; i < HDIM; ++i) { d1 += lam_q1[i] * lam_k1[i]; d2 += lam_q2[i] * lam_k2[i]; }
  const float lam = expf(d1) - expf(d2) + LAMBDA_INIT;

  const __bf16* Qrow = Qb + (size_t)(b * SS) * QK_LD;
  const __bf16* Krow = Kb + (size_t)(b * SS) * QK_LD;
  const __bf16* Vrow = Vb + (size_t)(b * SS) * DD;

  v16bf qf[2][2];
#pragma unroll
  for (int st = 0; st < 2; ++st)
#pragma unroll
    for (int c = 0; c < 2; ++c)
      qf[st][c] = load_bT_frag_h(Qrow, q0, h * 2 * HDIM + st * HDIM + c * 32, QK_LD);

  v8f o[2][4];
#pragma unroll
  for (int st = 0; st < 2; ++st)
#pragma unroll
    for (int f = 0; f < 4; ++f) o[st][f] = vzero8();
  float mrun[2] = {-1e30f, -1e30f};
  float lrun[2] = {0.f, 0.f};

  for (int k0 = 0; k0 < q0 + 16; k0 += 32) {
    v16bf vf[4];
#pragma unroll
    for (int f = 0; f < 4; ++f)
      vf[f] = load_b_frag_h(Vrow + (size_t)k0 * DD, 0, h * HDIM + f * 16, DD);

    float padv[2][8];
#pragma unroll
    for (int kh = 0; kh < 2; ++kh)
#pragma unroll
      for (int r = 0; r < 8; ++r)
        padv[kh][r] = (1.0f - amask[b * SS + k0 + kh * 16 + hi * 8 + r]) * -1e9f;

#pragma unroll
    for (int st = 0; st < 2; ++st) {
      v8f ct[2];
#pragma unroll
      for (int kh = 0; kh < 2; ++kh) {
        v8f c = vzero8();
#pragma unroll
        for (int cc = 0; cc < 2; ++cc) {
          v16bf ka = load_a_frag_h(Krow, k0 + kh * 16,
                                   h * 2 * HDIM + st * HDIM + cc * 32, QK_LD);
          c = wmma_bf16(ka, qf[st][cc], c);
        }
        ct[kh] = c;
      }
      const int q_abs = q0 + ln;
      float mloc = -1e30f;
      float sv[2][8];
#pragma unroll
      for (int kh = 0; kh < 2; ++kh)
#pragma unroll
        for (int r = 0; r < 8; ++r) {
          const int kv = k0 + kh * 16 + hi * 8 + r;
          float s = ct[kh][r] * scale - slope * (float)(q_abs - kv);
          if (kv > q_abs) s += -1e9f;
          s += padv[kh][r];
          sv[kh][r] = s;
          mloc = fmaxf(mloc, s);
        }
      mloc = fmaxf(mloc, __shfl_xor(mloc, 16, 32));
      const float mnew = fmaxf(mrun[st], mloc);
      const float corr = expf(mrun[st] - mnew);
      mrun[st] = mnew;

      float psum = 0.f;
      float pv[2][8];
#pragma unroll
      for (int kh = 0; kh < 2; ++kh)
#pragma unroll
        for (int r = 0; r < 8; ++r) {
          const float p = expf(sv[kh][r] - mnew);
          pv[kh][r] = p;
          psum += p;
        }
      psum += __shfl_xor(psum, 16, 32);
      lrun[st] = lrun[st] * corr + psum;

      float corrRow[8];
#pragma unroll
      for (int r = 0; r < 8; ++r) corrRow[r] = __shfl(corr, r + hi * 8, 32);
#pragma unroll
      for (int f = 0; f < 4; ++f)
#pragma unroll
        for (int r = 0; r < 8; ++r) o[st][f][r] *= corrRow[r];

      v16bf pa;
#pragma unroll
      for (int e = 0; e < 8; ++e) { pa[e] = (__bf16)pv[0][e]; pa[8 + e] = (__bf16)pv[1][e]; }
#pragma unroll
      for (int f = 0; f < 4; ++f) o[st][f] = wmma_bf16(pa, vf[f], o[st][f]);
    }
  }

  float lr1[8], lr2[8];
#pragma unroll
  for (int r = 0; r < 8; ++r) {
    lr1[r] = __shfl(lrun[0], r + hi * 8, 32);
    lr2[r] = __shfl(lrun[1], r + hi * 8, 32);
  }

  // sigma_q = elu(Q1)+1 as A-fragments; denom = sigma_q @ z (per q)
  v16bf sa[2];
  float zsum = 0.f;
#pragma unroll
  for (int c = 0; c < 2; ++c) {
    const __bf16* p = Qrow + (size_t)(q0 + ln) * QK_LD + h * 2 * HDIM + c * 32 + hi * 8;
    v16bf t;
#pragma unroll
    for (int e = 0; e < 8; ++e) {
      const int hd0 = c * 32 + hi * 8 + e;
      float x0 = (float)p[e];
      float s0 = ((x0 > 0.f) ? x0 : expm1f(x0)) + 1.0f;
      zsum += s0 * zv[h * HDIM + hd0];
      t[e] = (__bf16)s0;
      float x1 = (float)p[16 + e];
      float s1 = ((x1 > 0.f) ? x1 : expm1f(x1)) + 1.0f;
      zsum += s1 * zv[h * HDIM + hd0 + 16];
      t[8 + e] = (__bf16)s1;
    }
    sa[c] = t;
  }
  zsum += __shfl_xor(zsum, 16, 32);
  float zRow[8];
#pragma unroll
  for (int r = 0; r < 8; ++r) zRow[r] = __shfl(zsum, r + hi * 8, 32);

  v8f mv[4];
#pragma unroll
  for (int f = 0; f < 4; ++f) {
    v8f c = vzero8();
#pragma unroll
    for (int cc = 0; cc < 2; ++cc)
      c = wmma_bf16(sa[cc], load_b_frag_f(memM + (size_t)h * HDIM * HDIM, cc * 32, f * 16, HDIM), c);
    mv[f] = c;
  }

  float ssum = 0.f, ssq = 0.f;
#pragma unroll
  for (int f = 0; f < 4; ++f) {
    const float g = 1.0f / (1.0f + expf(-mgate[h * HDIM + f * 16 + ln]));
#pragma unroll
    for (int r = 0; r < 8; ++r) {
      const float att = o[0][f][r] / lr1[r] - lam * (o[1][f][r] / lr2[r]);
      const float x = g * (mv[f][r] / zRow[r]) + (1.0f - g) * att;
      const int qa = q0 + r + hi * 8;
      Xout[(size_t)(b * SS + qa) * DD + h * HDIM + f * 16 + ln] = x;
      ssum += x;
      ssq += x * x;
    }
  }
#pragma unroll
  for (int off = 16; off >= 1; off >>= 1) {
    ssum += __shfl_xor(ssum, off, 32);
    ssq += __shfl_xor(ssq, off, 32);
  }
  if (lane == 0) {
    atomicAdd(&stats[(b * HH + h) * 2 + 0], ssum);
    atomicAdd(&stats[(b * HH + h) * 2 + 1], ssq);
  }
}

// ---------------------------------------------------------------------------
__global__ void zero_kernel(float* p, int n) {
  int i = blockIdx.x * blockDim.x + threadIdx.x;
  if (i < n) p[i] = 0.f;
}

__global__ void f32_to_bf16_kernel(const float* __restrict__ x, __bf16* __restrict__ y,
                                   long n) {
  long i = (long)blockIdx.x * blockDim.x + threadIdx.x;
  if (i < n) y[i] = (__bf16)x[i];
}

__global__ void finalize_stats(const float* __restrict__ stats, float* __restrict__ mstd) {
  int i = threadIdx.x;
  if (i >= BB * HH) return;
  const float cnt = (float)(SS * HDIM);
  const float mean = stats[2 * i] / cnt;
  const float var = stats[2 * i + 1] / cnt - mean * mean;
  mstd[2 * i] = mean;
  mstd[2 * i + 1] = rsqrtf(var + 1e-5f);
}

__global__ void groupnorm_kernel(const float* __restrict__ X, const float* __restrict__ mstd,
                                 const float* __restrict__ gw, const float* __restrict__ gb,
                                 __bf16* __restrict__ Y) {
  size_t i = (size_t)blockIdx.x * blockDim.x + threadIdx.x;
  if (i >= (size_t)NROW * DD) return;
  const int d = (int)(i % DD);
  const int row = (int)(i / DD);
  const int bb = row / SS;
  const int hh = d / HDIM;
  const float mean = mstd[(bb * HH + hh) * 2];
  const float istd = mstd[(bb * HH + hh) * 2 + 1];
  Y[i] = (__bf16)(((X[i] - mean) * istd * gw[d] + gb[d]) * (1.0f - LAMBDA_INIT));
}

// ---------------------------------------------------------------------------
extern "C" void kernel_launch(void* const* d_in, const int* in_sizes, int n_in,
                              void* d_out, int out_size, void* d_ws, size_t ws_size,
                              hipStream_t stream) {
  const float* X      = (const float*)d_in[0];
  const float* amask  = (const float*)d_in[1];
  const float* Wq     = (const float*)d_in[2];
  const float* Wk     = (const float*)d_in[3];
  const float* Wv     = (const float*)d_in[4];
  const float* Wo     = (const float*)d_in[5];
  const float* lam_q1 = (const float*)d_in[6];
  const float* lam_q2 = (const float*)d_in[7];
  const float* lam_k1 = (const float*)d_in[8];
  const float* lam_k2 = (const float*)d_in[9];
  const float* gnw    = (const float*)d_in[10];
  const float* gnb    = (const float*)d_in[11];
  const float* mgate  = (const float*)d_in[12];
  const float* memM   = (const float*)d_in[13];
  const float* zv     = (const float*)d_in[14];
  float* out = (float*)d_out;

  // workspace layout (bf16-resident operands)
  char* w = (char*)d_ws;
  __bf16* Xh  = (__bf16*)w; w += (size_t)NROW * DD * 2;       // 8 MB
  __bf16* Wqh = (__bf16*)w; w += (size_t)DD * QK_LD * 2;      // 4 MB
  __bf16* Wkh = (__bf16*)w; w += (size_t)DD * QK_LD * 2;      // 4 MB
  __bf16* Wvh = (__bf16*)w; w += (size_t)DD * DD * 2;         // 2 MB
  __bf16* Woh = (__bf16*)w; w += (size_t)DD * DD * 2;         // 2 MB
  __bf16* Qh  = (__bf16*)w; w += (size_t)NROW * QK_LD * 2;    // 16 MB
  __bf16* Kh  = (__bf16*)w; w += (size_t)NROW * QK_LD * 2;    // 16 MB
  __bf16* Vh  = (__bf16*)w; w += (size_t)NROW * DD * 2;       // 8 MB
  float*  X1  = (float*)w;  w += (size_t)NROW * DD * 4;       // 16 MB (attn out)
  __bf16* Yh  = (__bf16*)w; w += (size_t)NROW * DD * 2;       // 8 MB (normalized)
  float*  stats = (float*)w; w += 64 * 4;
  float*  mstd  = (float*)w;

  zero_kernel<<<1, 128, 0, stream>>>(stats, 128);

  // one-time f32 -> bf16 downconversions
  const long nX = (long)NROW * DD, nWqk = (long)DD * QK_LD, nWsq = (long)DD * DD;
  f32_to_bf16_kernel<<<(int)((nX + 255) / 256), 256, 0, stream>>>(X, Xh, nX);
  f32_to_bf16_kernel<<<(int)((nWqk + 255) / 256), 256, 0, stream>>>(Wq, Wqh, nWqk);
  f32_to_bf16_kernel<<<(int)((nWqk + 255) / 256), 256, 0, stream>>>(Wk, Wkh, nWqk);
  f32_to_bf16_kernel<<<(int)((nWsq + 255) / 256), 256, 0, stream>>>(Wv, Wvh, nWsq);
  f32_to_bf16_kernel<<<(int)((nWsq + 255) / 256), 256, 0, stream>>>(Wo, Woh, nWsq);

  // QKV projections -> bf16 outputs (64x64 C tile per block, TDM-fed)
  gemm_wmma_h<true><<<dim3(QK_LD / 64, NROW / 64), 128, 0, stream>>>(Xh, Wqh, Qh, NROW, QK_LD, DD);
  gemm_wmma_h<true><<<dim3(QK_LD / 64, NROW / 64), 128, 0, stream>>>(Xh, Wkh, Kh, NROW, QK_LD, DD);
  gemm_wmma_h<true><<<dim3(DD / 64, NROW / 64), 128, 0, stream>>>(Xh, Wvh, Vh, NROW, DD, DD);

  // fused differential flash attention + memory term + stat accumulation
  diffattn_kernel<<<dim3(SS / 64, HH, BB), 128, 0, stream>>>(
      Qh, Kh, Vh, amask, lam_q1, lam_q2, lam_k1, lam_k2, mgate, memM, zv, X1, stats);

  finalize_stats<<<1, 32, 0, stream>>>(stats, mstd);

  groupnorm_kernel<<<(int)(((size_t)NROW * DD + 255) / 256), 256, 0, stream>>>(
      X1, mstd, gnw, gnb, Yh);

  // output projection (f32 result)
  gemm_wmma_h<false><<<dim3(DD / 64, NROW / 64), 128, 0, stream>>>(Yh, Woh, out, NROW, DD, DD);
}
